// MoELayer_17368847745265
// MI455X (gfx1250) — matrix operand order repaired and checked by reading, not compiled, not run on hardware
//
#include <hip/hip_runtime.h>
#include <hip/hip_bf16.h>

// ---------------------------------------------------------------------------
// MoE layer for gfx1250 (MI455X): fp32 router + capacity dispatch,
// bf16 WMMA grouped expert FFN (gelu(x@w1)@w2), fp32 combine.
// GEMM staging uses GLOBAL_LOAD_ASYNC_TO_LDS_B128 (ASYNCcnt path).
// ---------------------------------------------------------------------------

typedef __attribute__((ext_vector_type(16))) __bf16 v16bf;
typedef __attribute__((ext_vector_type(8)))  __bf16 v8bf;
typedef __attribute__((ext_vector_type(8)))  float  v8f;

#define NTOK 8192
#define HDIM 1024
#define IDIM 4096
#define NEXP 8
#define TOPK 2
#define CAP  2560          // ceil(1.25 * 8192 * 2 / 8)
#define NASSIGN (NTOK * TOPK)
#define NCHUNK 32
#define CHUNK (NASSIGN / NCHUNK)   // 512

#if defined(__gfx1250__) && __has_builtin(__builtin_amdgcn_global_load_async_to_lds_b128)
#define HAVE_ASYNC_LDS 1
#else
#define HAVE_ASYNC_LDS 0
#endif

// ---------------------------- Router ---------------------------------------
__global__ __launch_bounds__(256) void router_kernel(
    const float* __restrict__ tokens, const float* __restrict__ rw,
    int* __restrict__ expertId, float* __restrict__ gateW) {
  const int wid = threadIdx.x >> 5;
  const int lane = threadIdx.x & 31;
  const int t = blockIdx.x * 8 + wid;
  const float* x = tokens + (size_t)t * HDIM;

  float acc[NEXP];
#pragma unroll
  for (int e = 0; e < NEXP; ++e) acc[e] = 0.f;
  for (int h = lane; h < HDIM; h += 32) {
    const float xv = x[h];
#pragma unroll
    for (int e = 0; e < NEXP; ++e) acc[e] = fmaf(xv, rw[h * NEXP + e], acc[e]);
  }
#pragma unroll
  for (int e = 0; e < NEXP; ++e)
#pragma unroll
    for (int m = 16; m > 0; m >>= 1) acc[e] += __shfl_xor(acc[e], m, 32);

  if (lane == 0) {
    float mx = acc[0];
#pragma unroll
    for (int e = 1; e < NEXP; ++e) mx = fmaxf(mx, acc[e]);
    float p[NEXP], s = 0.f;
#pragma unroll
    for (int e = 0; e < NEXP; ++e) { p[e] = __expf(acc[e] - mx); s += p[e]; }
    const float invs = 1.f / s;
#pragma unroll
    for (int e = 0; e < NEXP; ++e) p[e] *= invs;
    int i1 = 0;
#pragma unroll
    for (int e = 1; e < NEXP; ++e) if (p[e] > p[i1]) i1 = e;
    int i2 = (i1 == 0) ? 1 : 0;
#pragma unroll
    for (int e = 0; e < NEXP; ++e) if (e != i1 && p[e] > p[i2]) i2 = e;
    const float inv = 1.f / (p[i1] + p[i2] + 1e-6f);
    expertId[2 * t + 0] = i1;
    expertId[2 * t + 1] = i2;
    gateW[2 * t + 0] = p[i1] * inv;
    gateW[2 * t + 1] = p[i2] * inv;
  }
}

// ------------------- Deterministic capacity dispatch ------------------------
__global__ void hist_kernel(const int* __restrict__ expertId, int* __restrict__ blkCnt) {
  __shared__ int cnt[NEXP];
  if (threadIdx.x < NEXP) cnt[threadIdx.x] = 0;
  __syncthreads();
  const int base = blockIdx.x * CHUNK;
  for (int j = threadIdx.x; j < CHUNK; j += blockDim.x)
    atomicAdd(&cnt[expertId[base + j]], 1);
  __syncthreads();
  if (threadIdx.x < NEXP) blkCnt[blockIdx.x * NEXP + threadIdx.x] = cnt[threadIdx.x];
}

__global__ void scan_kernel(const int* __restrict__ blkCnt, int* __restrict__ blkOff) {
  const int e = threadIdx.x;
  if (e >= NEXP) return;
  int run = 0;
  for (int b = 0; b < NCHUNK; ++b) {
    blkOff[b * NEXP + e] = run;
    run += blkCnt[b * NEXP + e];
  }
}

__global__ void assign_kernel(const int* __restrict__ expertId,
                              const int* __restrict__ blkOff,
                              int* __restrict__ slot, float* __restrict__ gateW) {
  const int b = threadIdx.x;
  if (b >= NCHUNK) return;
  int local[NEXP];
#pragma unroll
  for (int e = 0; e < NEXP; ++e) local[e] = blkOff[b * NEXP + e];
  for (int j = 0; j < CHUNK; ++j) {
    const int i = b * CHUNK + j;
    const int e = expertId[i];
    const int pos = local[e]++;
    if (pos < CAP) {
      slot[i] = e * CAP + pos;
    } else {
      slot[i] = -1;
      gateW[i] = 0.f;
    }
  }
}

__global__ void zero_kernel(unsigned int* __restrict__ p, size_t n) {
  size_t i = (size_t)blockIdx.x * blockDim.x + threadIdx.x;
  const size_t stride = (size_t)gridDim.x * blockDim.x;
  for (; i < n; i += stride) p[i] = 0u;
}

__global__ void scatter_kernel(const float* __restrict__ tokens,
                               const int* __restrict__ slot,
                               __bf16* __restrict__ disp) {
  const int i = blockIdx.x;
  const int s = slot[i];
  if (s < 0) return;
  const float* src = tokens + (size_t)(i >> 1) * HDIM;
  __bf16* d = disp + (size_t)s * HDIM;
  for (int j = threadIdx.x; j < HDIM; j += blockDim.x) d[j] = (__bf16)src[j];
}

// ----------------- fp32 -> bf16 transposed weight conversion ----------------
__global__ void transpose_bf16_kernel(const float* __restrict__ src,
                                      __bf16* __restrict__ dst, int R, int Ccols) {
  __shared__ float tile[32][33];
  const int e = blockIdx.z;
  const float* s = src + (size_t)e * R * Ccols;
  __bf16* d = dst + (size_t)e * R * Ccols;
  const int c0 = blockIdx.x * 32, r0 = blockIdx.y * 32;
  const int tx = threadIdx.x, ty = threadIdx.y;
  for (int y = ty; y < 32; y += 8)
    tile[y][tx] = s[(size_t)(r0 + y) * Ccols + c0 + tx];
  __syncthreads();
  for (int y = ty; y < 32; y += 8)
    d[(size_t)(c0 + y) * R + r0 + tx] = (__bf16)tile[tx][y];
}

// ----------------------------- WMMA GEMM ------------------------------------
__device__ __forceinline__ float gelu_tanh(float x) {
  const float k0 = 0.7978845608028654f;   // sqrt(2/pi)
  const float k1 = 0.044715f;
  return 0.5f * x * (1.0f + tanhf(k0 * (x + k1 * x * x * x)));
}

union Frag {
  v16bf v;
  v8bf h[2];
};

// Stage one 16B chunk global -> LDS. CDNA5 async-to-LDS path (ASYNCcnt,
// no VGPR staging); fall back to load+ds_store if builtin is absent.
__device__ __forceinline__ void stage16(const __bf16* g, __bf16* l) {
#if HAVE_ASYNC_LDS
  typedef int v4i32 __attribute__((vector_size(16)));
  __builtin_amdgcn_global_load_async_to_lds_b128(
      (__attribute__((address_space(1))) v4i32*)(g),
      (__attribute__((address_space(3))) v4i32*)(l), 0, 0);
#else
  *(uint4*)l = *(const uint4*)g;
#endif
}

__device__ __forceinline__ void stage_wait() {
#if HAVE_ASYNC_LDS
#if __has_builtin(__builtin_amdgcn_s_wait_asynccnt)
  __builtin_amdgcn_s_wait_asynccnt(0);
#else
  asm volatile("s_wait_asynccnt 0" ::: "memory");
#endif
#endif
}

#define LDS_PITCH 40   // 80 B rows: 16 B aligned, bank stride 20

// D(M x N) = A(M x K, row-major) * B^T with B stored N-major (N x K, K contig).
// Block tile 256x128, 8 waves (4x2), each wave computes 64x64 = 4x4 WMMA tiles.
// K-step 32. 16 WMMAs per 16 fragment loads per wave per K-step.
template <bool GELU, typename OutT>
__global__ __launch_bounds__(256) void moe_gemm_kernel(
    const __bf16* __restrict__ A, const __bf16* __restrict__ B,
    OutT* __restrict__ D, int M, int N, int K,
    long long strideA, long long strideB, long long strideD) {
  __shared__ __align__(16) __bf16 lA[256 * LDS_PITCH];
  __shared__ __align__(16) __bf16 lB[128 * LDS_PITCH];

  const int e = blockIdx.z;
  A += (size_t)e * strideA;
  B += (size_t)e * strideB;
  D += (size_t)e * strideD;

  const int tid = threadIdx.x;
  const int wid = tid >> 5, lane = tid & 31;
  const int waveM = wid >> 1;          // 0..3 -> 64-row slabs
  const int waveN = wid & 1;           // 0..1 -> 64-col slabs
  const int m0 = blockIdx.y * 256, n0 = blockIdx.x * 128;

  const int lrow = lane & 15;
  const int hi = lane >> 4;            // lane half
  const int kbA = hi << 3;             // A frag: K 0-7/16-23 vs 8-15/24-31
  const int kbB = hi << 3;             // B frag: K 0-7/16-23 vs 8-15/24-31

  v8f c[4][4];
  const v8f vzero = {0.f, 0.f, 0.f, 0.f, 0.f, 0.f, 0.f, 0.f};
#pragma unroll
  for (int sm = 0; sm < 4; ++sm)
#pragma unroll
    for (int sn = 0; sn < 4; ++sn) c[sm][sn] = vzero;

  for (int k0 = 0; k0 < K; k0 += 32) {
    const __bf16* gA = A + (size_t)m0 * K + k0;
    const __bf16* gB = B + (size_t)n0 * K + k0;
    // A tile: 256 rows x 32 cols = 1024 chunks of 16 B; 4 per thread.
#pragma unroll
    for (int it = 0; it < 4; ++it) {
      const int cidx = tid + it * 256;
      const int row = cidx >> 2, q = cidx & 3;
      stage16(gA + (size_t)row * K + q * 8, &lA[row * LDS_PITCH + q * 8]);
    }
    // B tile: 128 rows x 32 cols = 512 chunks; 2 per thread.
#pragma unroll
    for (int it = 0; it < 2; ++it) {
      const int cidx = tid + it * 256;
      const int row = cidx >> 2, q = cidx & 3;
      stage16(gB + (size_t)row * K + q * 8, &lB[row * LDS_PITCH + q * 8]);
    }
    __builtin_prefetch(gA + (size_t)(tid) * K + 32, 0, 1);
    __builtin_prefetch(gB + (size_t)(tid >> 1) * K + 32, 0, 1);
    stage_wait();
    __syncthreads();

    Frag aF[4], bF[4];
#pragma unroll
    for (int sm = 0; sm < 4; ++sm) {
      const int r = waveM * 64 + sm * 16 + lrow;
      aF[sm].h[0] = *(const v8bf*)&lA[r * LDS_PITCH + kbA];
      aF[sm].h[1] = *(const v8bf*)&lA[r * LDS_PITCH + kbA + 16];
    }
#pragma unroll
    for (int sn = 0; sn < 4; ++sn) {
      const int r = waveN * 64 + sn * 16 + lrow;
      bF[sn].h[0] = *(const v8bf*)&lB[r * LDS_PITCH + kbB];
      bF[sn].h[1] = *(const v8bf*)&lB[r * LDS_PITCH + kbB + 16];
    }
#pragma unroll
    for (int sm = 0; sm < 4; ++sm)
#pragma unroll
      for (int sn = 0; sn < 4; ++sn)
        c[sm][sn] = __builtin_amdgcn_wmma_f32_16x16x32_bf16(
            false, aF[sm].v, false, bF[sn].v, (short)0, c[sm][sn], false, false);
    __syncthreads();
  }

  // Epilogue: VGPR r holds M=r (lanes 0-15) / M=r+8 (lanes 16-31).
  const int rbase = hi << 3;
#pragma unroll
  for (int sm = 0; sm < 4; ++sm) {
#pragma unroll
    for (int sn = 0; sn < 4; ++sn) {
      const int col = n0 + waveN * 64 + sn * 16 + lrow;
#pragma unroll
      for (int r = 0; r < 8; ++r) {
        const int row = m0 + waveM * 64 + sm * 16 + rbase + r;
        float v = c[sm][sn][r];
        if (GELU) v = gelu_tanh(v);
        D[(size_t)row * N + col] = (OutT)v;
      }
    }
  }
}

// ----------------------------- Combine --------------------------------------
__global__ __launch_bounds__(256) void combine_kernel(
    const float* __restrict__ eout, const int* __restrict__ slot,
    const float* __restrict__ gateW, float* __restrict__ out) {
  const int n = blockIdx.x;
  const int h = threadIdx.x * 4;
  float acc[4] = {0.f, 0.f, 0.f, 0.f};
#pragma unroll
  for (int k = 0; k < TOPK; ++k) {
    const int i = n * TOPK + k;
    const int s = slot[i];
    const float w = gateW[i];
    if (s >= 0) {
      const float* src = eout + (size_t)s * HDIM + h;
#pragma unroll
      for (int j = 0; j < 4; ++j) acc[j] += w * src[j];
    }
  }
  float* o = out + (size_t)n * HDIM + h;
#pragma unroll
  for (int j = 0; j < 4; ++j) o[j] = acc[j];
}

// ----------------------------- Launch ---------------------------------------
extern "C" void kernel_launch(void* const* d_in, const int* in_sizes, int n_in,
                              void* d_out, int out_size, void* d_ws, size_t ws_size,
                              hipStream_t stream) {
  const float* tokens = (const float*)d_in[0];
  const float* router_weight = (const float*)d_in[1];
  const float* w1 = (const float*)d_in[2];
  const float* w2 = (const float*)d_in[3];
  float* out = (float*)d_out;

  char* ws = (char*)d_ws;
  size_t off = 0;
  auto alloc = [&](size_t bytes) -> void* {
    void* p = ws + off;
    off += (bytes + 255) & ~(size_t)255;
    return p;
  };

  int* expertId = (int*)alloc((size_t)NASSIGN * 4);
  float* gateW  = (float*)alloc((size_t)NASSIGN * 4);
  int* slot     = (int*)alloc((size_t)NASSIGN * 4);
  int* blkCnt   = (int*)alloc((size_t)NCHUNK * NEXP * 4);
  int* blkOff   = (int*)alloc((size_t)NCHUNK * NEXP * 4);
  __bf16* w1t   = (__bf16*)alloc((size_t)NEXP * HDIM * IDIM * 2);
  __bf16* w2t   = (__bf16*)alloc((size_t)NEXP * HDIM * IDIM * 2);
  __bf16* disp  = (__bf16*)alloc((size_t)NEXP * CAP * HDIM * 2);
  __bf16* hidden= (__bf16*)alloc((size_t)NEXP * CAP * IDIM * 2);
  float* eout   = (float*)alloc((size_t)NEXP * CAP * HDIM * 4);

  // 1) Router
  router_kernel<<<NTOK / 8, 256, 0, stream>>>(tokens, router_weight, expertId, gateW);

  // 2) Capacity dispatch bookkeeping (deterministic, order-preserving)
  hist_kernel<<<NCHUNK, 256, 0, stream>>>(expertId, blkCnt);
  scan_kernel<<<1, NEXP, 0, stream>>>(blkCnt, blkOff);
  assign_kernel<<<1, NCHUNK, 0, stream>>>(expertId, blkOff, slot, gateW);

  // 3) Weight conversion (fp32 -> bf16, transposed so B operands are K-contig)
  dim3 tb(32, 8);
  transpose_bf16_kernel<<<dim3(IDIM / 32, HDIM / 32, NEXP), tb, 0, stream>>>(
      w1, w1t, HDIM, IDIM);   // (E,H,I) -> (E,I,H)
  transpose_bf16_kernel<<<dim3(HDIM / 32, IDIM / 32, NEXP), tb, 0, stream>>>(
      w2, w2t, IDIM, HDIM);   // (E,I,H) -> (E,H,I)

  // 4) Zero + scatter tokens into per-expert capacity buffers (bf16)
  zero_kernel<<<2048, 256, 0, stream>>>((unsigned int*)disp,
                                        (size_t)NEXP * CAP * HDIM / 2);
  scatter_kernel<<<NASSIGN, 256, 0, stream>>>(tokens, slot, disp);

  // 5) Expert FFN: hidden = gelu(disp @ w1) ; eout = hidden @ w2
  moe_gemm_kernel<true, __bf16><<<dim3(IDIM / 128, CAP / 256, NEXP), 256, 0, stream>>>(
      disp, w1t, hidden, CAP, IDIM, HDIM,
      (long long)CAP * HDIM, (long long)IDIM * HDIM, (long long)CAP * IDIM);
  moe_gemm_kernel<false, float><<<dim3(HDIM / 128, CAP / 256, NEXP), 256, 0, stream>>>(
      hidden, w2t, eout, CAP, HDIM, IDIM,
      (long long)CAP * IDIM, (long long)HDIM * IDIM, (long long)CAP * HDIM);

  // 6) Combine
  combine_kernel<<<NTOK, 256, 0, stream>>>(eout, slot, gateW, out);
}
